// Qwen3Attention_48876727829004
// MI455X (gfx1250) — matrix operand order repaired and checked by reading, not compiled, not run on hardware
//
#include <hip/hip_runtime.h>

// ---------------------------------------------------------------------------
// Qwen3-style attention block for MI455X (gfx1250, wave32, WMMA 16x16x32 f16)
// B=2, T=2048, D=2048, H=16, KH=4, HD=128 (GQA 4:1), causal, RMSNorm+RoPE.
// Data movement: async global->LDS (ASYNCcnt) with LDS double buffering.
// ---------------------------------------------------------------------------

typedef __attribute__((ext_vector_type(16))) _Float16 v16h;
typedef __attribute__((ext_vector_type(8)))  float    v8f;
typedef __attribute__((ext_vector_type(4)))  int      i32x4;

#define B_   2
#define T_   2048
#define D_   2048
#define H_   16
#define KH_  4
#define HD_  128

#if __has_builtin(__builtin_amdgcn_global_load_async_to_lds_b128) && \
    __has_builtin(__builtin_amdgcn_s_wait_asynccnt)
#define HAVE_ASYNC_LDS 1
#endif

// 16-byte global -> LDS copy (async DMA when available, sync fallback).
__device__ __forceinline__ void cp16_g2l(void* lds, const void* g) {
#ifdef HAVE_ASYNC_LDS
  __builtin_amdgcn_global_load_async_to_lds_b128(
      (__attribute__((address_space(1))) i32x4*)(void*)(g),
      (__attribute__((address_space(3))) i32x4*)(lds),
      /*imm offset=*/0, /*cpol=*/0);
#else
  *reinterpret_cast<uint4*>(lds) = *reinterpret_cast<const uint4*>(g);
#endif
}

__device__ __forceinline__ void cp_wait_all() {
#ifdef HAVE_ASYNC_LDS
  __builtin_amdgcn_s_wait_asynccnt(0);
#endif
}

// ---- WMMA helpers ----------------------------------------------------------

__device__ __forceinline__ v8f wmma16(v16h a, v16h b, v8f c) {
  // D = A(16x32 f16) * B(32x16 f16) + C(16x16 f32)
  return __builtin_amdgcn_wmma_f32_16x16x32_f16(
      /*neg_a=*/false, a, /*neg_b=*/false, b,
      /*c_mod=*/(short)0, c, /*reuse_a=*/false, /*reuse_b=*/false);
}

// A-matrix 16x32 f16 fragment from row-major LDS tile (ldh in halves).
// ISA layout: lane L (&15) = row M; lanes<16 hold K {0..7,16..23},
// lanes>=16 hold K {8..15,24..31}  -> two contiguous 16B reads per lane.
__device__ __forceinline__ v16h frag_ld_a(const _Float16* base, int ldh, int lane) {
  int m    = lane & 15;
  int koff = (lane >> 4) << 3;           // 0 or 8
  const _Float16* p = base + (size_t)m * ldh;
  union { v16h v; uint4 u[2]; } r;
  r.u[0] = *reinterpret_cast<const uint4*>(p + koff);        // K = koff..koff+7
  r.u[1] = *reinterpret_cast<const uint4*>(p + 16 + koff);   // K = 16+koff..
  return r.v;
}

// B-matrix 32x16 f16 fragment; tile stored as [n][k] row-major in LDS.
// lane&15 = column N; lanes<16 hold K {0..7,16..23}, lanes>=16 hold K {8..15,24..31}
// -> two contiguous 16B reads per lane.
__device__ __forceinline__ v16h frag_ld_b(const _Float16* base, int ldh, int lane) {
  int n    = lane & 15;
  int koff = (lane >> 4) << 3;           // 0 or 8
  const _Float16* p = base + (size_t)n * ldh;
  union { v16h v; uint4 u[2]; } r;
  r.u[0] = *reinterpret_cast<const uint4*>(p + koff);
  r.u[1] = *reinterpret_cast<const uint4*>(p + 16 + koff);
  return r.v;
}

// ---- elementwise f32 -> f16 ------------------------------------------------

__global__ void cvt_f32_f16(const float* __restrict__ in,
                            _Float16* __restrict__ out, size_t n) {
  size_t i = ((size_t)blockIdx.x * blockDim.x + threadIdx.x) * 4;
  if (i + 3 < n) {
    float4 v = *reinterpret_cast<const float4*>(in + i);
    out[i + 0] = (_Float16)v.x;
    out[i + 1] = (_Float16)v.y;
    out[i + 2] = (_Float16)v.z;
    out[i + 3] = (_Float16)v.w;
  }
}

// ---- WMMA GEMM: C[M,N] = A[M,K] * W[N,K]^T  (A,W f16 row-major, C f32) -----
// 128x128 tile per block, 8 waves, each wave = 32x64 (2x4 WMMA accumulators).
// Async global->LDS fills, double-buffered: one barrier per K-step.

__global__ void __launch_bounds__(256)
gemm_f16(const _Float16* __restrict__ A, const _Float16* __restrict__ W,
         float* __restrict__ C, int M, int N, int K) {
  __shared__ _Float16 As[2][128][40];   // 32 K-halves + 8 pad (16B-aligned rows)
  __shared__ _Float16 Bs[2][128][40];

  const int tid  = threadIdx.x;
  const int lane = tid & 31;
  const int wave = tid >> 5;
  const int wm   = (wave & 3) * 32;       // wave row offset in tile
  const int wn   = (wave >> 2) * 64;      // wave col offset in tile
  const int m0   = blockIdx.y * 128;
  const int n0   = blockIdx.x * 128;

  v8f acc[2][4] = {};

  const int r   = tid >> 1;               // tile row this thread fills
  const int off = (tid & 1) * 16;         // 16 halves per thread per matrix

  const _Float16* arow = A + (size_t)(m0 + r) * K + off;
  const _Float16* brow = W + (size_t)(n0 + r) * K + off;

  auto fill = [&](int buf, int k0) {
    cp16_g2l(&As[buf][r][off],     arow + k0);
    cp16_g2l(&As[buf][r][off + 8], arow + k0 + 8);
    cp16_g2l(&Bs[buf][r][off],     brow + k0);
    cp16_g2l(&Bs[buf][r][off + 8], brow + k0 + 8);
  };

  fill(0, 0);

  for (int k0 = 0; k0 < K; k0 += 32) {
    const int buf = (k0 >> 5) & 1;
    cp_wait_all();        // my async fills of `buf` have landed in LDS
    __syncthreads();      // everyone's fills landed; prev reads of buf^1 done

    if (k0 + 32 < K) {
      fill(buf ^ 1, k0 + 32);                       // overlap DMA with WMMA
      __builtin_prefetch(arow + k0 + 64, 0, 1);     // 2 K-steps lookahead
      __builtin_prefetch(brow + k0 + 64, 0, 1);
    }

    v16h af0 = frag_ld_a(&As[buf][wm][0],      40, lane);
    v16h af1 = frag_ld_a(&As[buf][wm + 16][0], 40, lane);
#pragma unroll
    for (int j = 0; j < 4; ++j) {
      v16h bf = frag_ld_b(&Bs[buf][wn + j * 16][0], 40, lane);
      acc[0][j] = wmma16(af0, bf, acc[0][j]);
      acc[1][j] = wmma16(af1, bf, acc[1][j]);
    }
  }

  const int ln  = lane & 15;
  const int hi8 = (lane >> 4) << 3;       // C-layout: lanes>=16 hold rows M+8
#pragma unroll
  for (int i = 0; i < 2; ++i)
#pragma unroll
    for (int j = 0; j < 4; ++j)
#pragma unroll
      for (int rr = 0; rr < 8; ++rr) {
        int m = m0 + wm + i * 16 + hi8 + rr;
        int n = n0 + wn + j * 16 + ln;
        C[(size_t)m * N + n] = acc[i][j][rr];
      }
}

// ---- RMSNorm + RoPE, f32 [bt, nh*HD] -> f16 [(b*nh+h)*T + t][HD] -----------

__global__ void __launch_bounds__(128)
norm_rope(const float* __restrict__ in, const float* __restrict__ w,
          const float* __restrict__ cosb, const float* __restrict__ sinb,
          _Float16* __restrict__ out, int nh) {
  const int bt = blockIdx.x;              // 0 .. B*T-1
  const int h  = blockIdx.y;
  const int d  = threadIdx.x;             // 0 .. 127
  const int b  = bt / T_;
  const int t  = bt - b * T_;

  __shared__ float red[HD_];
  __shared__ float xs[HD_];

  float x = in[(size_t)bt * (nh * HD_) + h * HD_ + d];
  red[d] = x * x;
  __syncthreads();
#pragma unroll
  for (int s = 64; s > 0; s >>= 1) {
    if (d < s) red[d] += red[d + s];
    __syncthreads();
  }
  float rms = rsqrtf(red[0] * (1.0f / HD_) + 1e-6f);
  xs[d] = x * rms * w[d];
  __syncthreads();

  const int i = d >> 1;
  float c = cosb[(size_t)t * (HD_ / 2) + i];
  float s = sinb[(size_t)t * (HD_ / 2) + i];
  float o = ((d & 1) == 0) ? (xs[d] * c - xs[d + 1] * s)
                           : (xs[d - 1] * s + xs[d] * c);
  out[((size_t)(b * nh + h) * T_ + t) * HD_ + d] = (_Float16)o;
}

// ---- V: f32 [bt, KH*HD] -> f16 transposed [(b*KH+kh)*HD + d][T] ------------

__global__ void v_cvt_t(const float* __restrict__ v, _Float16* __restrict__ vt) {
  size_t idx = (size_t)blockIdx.x * blockDim.x + threadIdx.x;  // over B*T*KH*HD
  int d  = (int)(idx & (HD_ - 1));
  size_t rr = idx >> 7;                   // /HD_
  int kh = (int)(rr & (KH_ - 1));
  size_t bt = rr >> 2;                    // /KH_
  int t = (int)(bt & (T_ - 1));
  int b = (int)(bt >> 11);                // /T_
  vt[((size_t)(b * KH_ + kh) * HD_ + d) * T_ + t] = (_Float16)v[idx];
}

// ---- Flash attention (causal, GQA 4:1) -------------------------------------
// Block = (b, h, 128-row Q tile), 8 waves * 16 rows. KV streamed 32 at a time.

__global__ void __launch_bounds__(256)
flash_attn(const _Float16* __restrict__ Q,   // [(b*H+h)*T + t][HD]
           const _Float16* __restrict__ Kg,  // [(b*KH+kh)*T + s][HD]
           const _Float16* __restrict__ Vt,  // [(b*KH+kh)*HD + d][T]
           float* __restrict__ O) {          // [b*T + t][H*HD]
  __shared__ _Float16 Qs[128][HD_];          // 32 KB
  __shared__ _Float16 Ks[32][HD_];           //  8 KB
  __shared__ _Float16 Vs[HD_][32];           //  8 KB  (V^T tile: [d][s])
  __shared__ _Float16 Ps[8][16][32];         //  8 KB  (per-wave P staging)

  const int tid  = threadIdx.x;
  const int lane = tid & 31;
  const int wave = tid >> 5;
  const int ln   = lane & 15;
  const int hi8  = (lane >> 4) << 3;

  const int qb = blockIdx.x, h = blockIdx.y, b = blockIdx.z;
  const int kh = h >> 2;                  // H/KH = 4
  const int qbase = qb * 128;

  // Async-fill Q tile (128 x 128 halves): 64 halves per thread.
  {
    const _Float16* qg = Q + ((size_t)(b * H_ + h) * T_ + qbase) * HD_;
    int r = tid >> 1, off = (tid & 1) * 64;
#pragma unroll
    for (int u = 0; u < 8; ++u)
      cp16_g2l(&Qs[r][off + u * 8], &qg[(size_t)r * HD_ + off + u * 8]);
  }

  float mrun[8], lrun[8];
  v8f oacc[8] = {};
#pragma unroll
  for (int rr = 0; rr < 8; ++rr) { mrun[rr] = -1e30f; lrun[rr] = 0.0f; }

  const float scale = 0.08838834764831845f;   // 1/sqrt(128)
  const int jmax = (qbase + 127) >> 5;

  const _Float16* kbase = Kg + (size_t)(b * KH_ + kh) * T_ * HD_;
  const _Float16* vbase = Vt + (size_t)(b * KH_ + kh) * HD_ * T_;

  for (int j = 0; j <= jmax; ++j) {
    const int s0 = j * 32;
    {   // K tile 32 x 128, V^T tile 128 x 32: 16 halves per thread each.
      int kr = tid >> 3, koff = (tid & 7) * 16;
      const _Float16* kg = kbase + (size_t)(s0 + kr) * HD_ + koff;
      cp16_g2l(&Ks[kr][koff],     kg);
      cp16_g2l(&Ks[kr][koff + 8], kg + 8);
      int dr = tid >> 1, voff = (tid & 1) * 16;
      const _Float16* vg = vbase + (size_t)dr * T_ + s0 + voff;
      cp16_g2l(&Vs[dr][voff],     vg);
      cp16_g2l(&Vs[dr][voff + 8], vg + 8);
    }
    cp_wait_all();
    __syncthreads();

    // S = Q K^T for this wave's 16 rows x 32 cols.
    v8f sacc[2] = {};
#pragma unroll
    for (int c = 0; c < 4; ++c) {
      v16h aq  = frag_ld_a(&Qs[wave * 16][c * 32], HD_, lane);
      v16h bk0 = frag_ld_b(&Ks[0][c * 32],  HD_, lane);
      v16h bk1 = frag_ld_b(&Ks[16][c * 32], HD_, lane);
      sacc[0] = wmma16(aq, bk0, sacc[0]);
      sacc[1] = wmma16(aq, bk1, sacc[1]);
    }

    // scale + causal mask + online softmax (rows live in 16-lane halves).
#pragma unroll
    for (int rr = 0; rr < 8; ++rr) {
      const int qi = qbase + wave * 16 + hi8 + rr;
      float sv0 = sacc[0][rr] * scale;
      float sv1 = sacc[1][rr] * scale;
      if (s0 + ln      > qi) sv0 = -1e30f;
      if (s0 + 16 + ln > qi) sv1 = -1e30f;

      float mx = fmaxf(sv0, sv1);
#pragma unroll
      for (int o = 8; o >= 1; o >>= 1) mx = fmaxf(mx, __shfl_xor(mx, o, 32));
      float mnew  = fmaxf(mrun[rr], mx);
      float alpha = __expf(mrun[rr] - mnew);
      mrun[rr] = mnew;

      float p0 = __expf(sv0 - mnew);
      float p1 = __expf(sv1 - mnew);
      float rs = p0 + p1;
#pragma unroll
      for (int o = 8; o >= 1; o >>= 1) rs += __shfl_xor(rs, o, 32);
      lrun[rr] = lrun[rr] * alpha + rs;
#pragma unroll
      for (int dt = 0; dt < 8; ++dt) oacc[dt][rr] *= alpha;

      Ps[wave][hi8 + rr][ln]      = (_Float16)p0;
      Ps[wave][hi8 + rr][16 + ln] = (_Float16)p1;
    }
    __syncthreads();   // P visible (ds-wait) + all waves done with K tile

    // O += P (16x32) * V (32x128)
    v16h ap = frag_ld_a(&Ps[wave][0][0], 32, lane);
#pragma unroll
    for (int dt = 0; dt < 8; ++dt) {
      v16h bv = frag_ld_b(&Vs[dt * 16][0], 32, lane);
      oacc[dt] = wmma16(ap, bv, oacc[dt]);
    }
    __syncthreads();   // before next KV tile overwrite
  }

  // Normalize + write out: O[b*T+t][h*HD + d], f32.
#pragma unroll
  for (int rr = 0; rr < 8; ++rr) {
    const int t = qbase + wave * 16 + hi8 + rr;
    const float inv = 1.0f / lrun[rr];
#pragma unroll
    for (int dt = 0; dt < 8; ++dt) {
      int d = dt * 16 + ln;
      O[((size_t)b * T_ + t) * (H_ * HD_) + h * HD_ + d] = oacc[dt][rr] * inv;
    }
  }
}

// ---------------------------------------------------------------------------

extern "C" void kernel_launch(void* const* d_in, const int* in_sizes, int n_in,
                              void* d_out, int out_size, void* d_ws, size_t ws_size,
                              hipStream_t stream) {
  const float* x    = (const float*)d_in[0];
  const float* Wq   = (const float*)d_in[1];
  const float* Wk   = (const float*)d_in[2];
  const float* Wv   = (const float*)d_in[3];
  const float* Wo   = (const float*)d_in[4];
  const float* qnw  = (const float*)d_in[5];
  const float* knw  = (const float*)d_in[6];
  const float* rc   = (const float*)d_in[7];
  const float* rs   = (const float*)d_in[8];
  // d_in[9] = mask (implemented analytically as causal)

  char* ws = (char*)d_ws;
  const size_t MB = (size_t)1 << 20;
  _Float16* x16   = (_Float16*)(ws + 0 * MB);      // 16 MB (reused as attn16)
  _Float16* wq16  = (_Float16*)(ws + 16 * MB);     //  8 MB
  _Float16* wk16  = (_Float16*)(ws + 24 * MB);     //  2 MB
  _Float16* wv16  = (_Float16*)(ws + 26 * MB);     //  2 MB
  _Float16* wo16  = (_Float16*)(ws + 28 * MB);     //  8 MB
  float*    q32   = (float*)   (ws + 36 * MB);     // 32 MB (reused as attn32)
  float*    k32   = (float*)   (ws + 68 * MB);     //  8 MB
  float*    v32   = (float*)   (ws + 76 * MB);     //  8 MB
  _Float16* q16   = (_Float16*)(ws + 84 * MB);     // 16 MB
  _Float16* k16   = (_Float16*)(ws + 100 * MB);    //  4 MB
  _Float16* vt16  = (_Float16*)(ws + 104 * MB);    //  4 MB
  float*    attn32 = q32;
  _Float16* attn16 = x16;

  const size_t nX  = (size_t)B_ * T_ * D_;         // 8388608
  const size_t nWq = (size_t)D_ * D_;              // 4194304
  const size_t nWk = (size_t)KH_ * HD_ * D_;       // 1048576

  // 1) f32 -> f16 conversions
  cvt_f32_f16<<<(unsigned)(nX  / 1024), 256, 0, stream>>>(x,  x16,  nX);
  cvt_f32_f16<<<(unsigned)(nWq / 1024), 256, 0, stream>>>(Wq, wq16, nWq);
  cvt_f32_f16<<<(unsigned)(nWk / 1024), 256, 0, stream>>>(Wk, wk16, nWk);
  cvt_f32_f16<<<(unsigned)(nWk / 1024), 256, 0, stream>>>(Wv, wv16, nWk);
  cvt_f32_f16<<<(unsigned)(nWq / 1024), 256, 0, stream>>>(Wo, wo16, nWq);

  // 2) QKV projections (WMMA GEMM): M = B*T = 4096
  const int M = B_ * T_;
  gemm_f16<<<dim3(D_ / 128,          M / 128), 256, 0, stream>>>(x16, wq16, q32, M, D_,        D_);
  gemm_f16<<<dim3((KH_*HD_) / 128,   M / 128), 256, 0, stream>>>(x16, wk16, k32, M, KH_ * HD_, D_);
  gemm_f16<<<dim3((KH_*HD_) / 128,   M / 128), 256, 0, stream>>>(x16, wv16, v32, M, KH_ * HD_, D_);

  // 3) RMSNorm + RoPE -> f16 head-major Q/K; V -> f16 transposed per head
  norm_rope<<<dim3(M, H_),  128, 0, stream>>>(q32, qnw, rc, rs, q16, H_);
  norm_rope<<<dim3(M, KH_), 128, 0, stream>>>(k32, knw, rc, rs, k16, KH_);
  v_cvt_t<<<(unsigned)(((size_t)M * KH_ * HD_) / 256), 256, 0, stream>>>(v32, vt16);

  // 4) Flash attention (WMMA QK^T and PV, online softmax)
  flash_attn<<<dim3(T_ / 128, H_, B_), 256, 0, stream>>>(q16, k16, vt16, attn32);

  // 5) Output projection
  cvt_f32_f16<<<(unsigned)(nX / 1024), 256, 0, stream>>>(attn32, attn16, nX);
  gemm_f16<<<dim3(D_ / 128, M / 128), 256, 0, stream>>>(attn16, wo16, (float*)d_out,
                                                        M, D_, D_);
}
